// MLA_41523743818316
// MI455X (gfx1250) — matrix operand order repaired and checked by reading, not compiled, not run on hardware
//
#include <hip/hip_runtime.h>
#include <hip/hip_bf16.h>

// ---------------------------------------------------------------------------
// MLA forward for MI455X (gfx1250, wave32, WMMA + Tensor Data Mover).
// All operands pre-split once into packed bf16 hi/lo (bf16x3 trick =>
// near-fp32 accuracy, fp32 accumulation). GEMM tiles are DMA'd to LDS by the
// TDM (tensor_load_to_lds) with descriptor-side padding, double-buffered and
// synchronized with s_wait_tensorcnt.
// ---------------------------------------------------------------------------

typedef __bf16 bf16_t;
typedef __attribute__((ext_vector_type(16))) __bf16 v16bf;
typedef __attribute__((ext_vector_type(8)))  __bf16 v8bf;
typedef __attribute__((ext_vector_type(4)))  __bf16 v4bf;
typedef __attribute__((ext_vector_type(8)))  float  v8f;
typedef __attribute__((ext_vector_type(4)))  float  f32x4;
typedef __attribute__((ext_vector_type(4)))  unsigned int v4u;
typedef __attribute__((ext_vector_type(8)))  int v8i;
typedef __attribute__((ext_vector_type(4)))  int v4i;

#define DIMC 2048
#define NH   16
#define HD   128
#define KVR  512
#define SEQ  2048

#if __has_builtin(__builtin_amdgcn_tensor_load_to_lds) && \
    __has_builtin(__builtin_amdgcn_s_wait_tensorcnt)
#define MLA_USE_TDM 1
#else
#define MLA_USE_TDM 0
#endif

// Exact truncating split: x == hi + r exactly; lo = trunc_bf16(r).
__device__ __forceinline__ void splitbf(float x, bf16_t& h, bf16_t& l) {
  unsigned u = __float_as_uint(x);
  h = __builtin_bit_cast(bf16_t, (unsigned short)(u >> 16));
  float r = x - __uint_as_float(u & 0xFFFF0000u);
  l = __builtin_bit_cast(bf16_t, (unsigned short)(__float_as_uint(r) >> 16));
}

__device__ __forceinline__ v16bf cat8(v8bf a, v8bf b) {
  return __builtin_shufflevector(a, b, 0,1,2,3,4,5,6,7,8,9,10,11,12,13,14,15);
}

__device__ __forceinline__ v8f wmma_bf(v16bf a, v16bf b, v8f c) {
  return __builtin_amdgcn_wmma_f32_16x16x32_bf16(false, a, false, b, (short)0, c,
                                                 false, false);
}

__device__ __forceinline__ float rmax16(float v) {
  v = fmaxf(v, __shfl_xor(v, 1, 32));
  v = fmaxf(v, __shfl_xor(v, 2, 32));
  v = fmaxf(v, __shfl_xor(v, 4, 32));
  v = fmaxf(v, __shfl_xor(v, 8, 32));
  return v;
}
__device__ __forceinline__ float rsum16(float v) {
  v += __shfl_xor(v, 1, 32);
  v += __shfl_xor(v, 2, 32);
  v += __shfl_xor(v, 4, 32);
  v += __shfl_xor(v, 8, 32);
  return v;
}

__device__ __forceinline__ void wave_lds_sync() {
#if __has_builtin(__builtin_amdgcn_fence)
  __builtin_amdgcn_fence(__ATOMIC_RELEASE, "wavefront");
#endif
  __builtin_amdgcn_wave_barrier();
#if __has_builtin(__builtin_amdgcn_fence)
  __builtin_amdgcn_fence(__ATOMIC_ACQUIRE, "wavefront");
#endif
}

// ---------------------------------------------------------------------------
// fp32 -> packed bf16 hi/lo split pass (one thread per float4 chunk).
// ---------------------------------------------------------------------------
__global__ __launch_bounds__(256)
void mla_split(const float* __restrict__ s, bf16_t* __restrict__ h,
               bf16_t* __restrict__ l, int n4) {
  int i = blockIdx.x * 256 + threadIdx.x;
  if (i >= n4) return;
  f32x4 v = *(const f32x4*)(s + (size_t)i * 4);
  v4bf hv, lv;
  #pragma unroll
  for (int e = 0; e < 4; ++e) {
    bf16_t hb, lb; splitbf(v[e], hb, lb);
    hv[e] = hb; lv[e] = lb;
  }
  *(v4bf*)(h + (size_t)i * 4) = hv;
  *(v4bf*)(l + (size_t)i * 4) = lv;
}

// ---------------------------------------------------------------------------
// TDM: load a 2D bf16 tile (tileW=32 elems x rows) from global (row stride
// strideElems) into LDS at lds byte offset, inserting 16B pad per 64B row so
// the LDS row stride is GPAD(=40) elems. D# built per CDNA5 ISA ch.8.
// ---------------------------------------------------------------------------
#define GTM 128
#define GTN 64
#define GTK 32
#define GPAD 40

#if MLA_USE_TDM
__device__ __forceinline__ void tdm_load_tile(const bf16_t* ldsDst,
                                              const bf16_t* gsrc,
                                              unsigned rows,
                                              unsigned strideElems) {
  unsigned long long ga = (unsigned long long)(uintptr_t)gsrc;
  v4u g0;
  g0[0] = 1u;                                        // count=1, user mode
  g0[1] = (unsigned)(uintptr_t)ldsDst;               // lds_addr (bytes)
  g0[2] = (unsigned)ga;                              // global_addr[31:0]
  g0[3] = ((unsigned)(ga >> 32) & 0x01ffffffu) | (2u << 30);  // ga[56:32], type=2
  v8i g1;
  // data_size=1(2B) | pad_enable | pad_interval=3(16 DW) | pad_amount=3(4 DW)
  g1[0] = (int)((1u << 16) | (1u << 20) | (3u << 22) | (3u << 25));
  g1[1] = (int)((unsigned)GTK << 16);                // tensor_dim0[15:0]=32
  g1[2] = (int)(rows << 16);                         // tdim0 hi=0; tensor_dim1=rows
  g1[3] = (int)((unsigned)GTK << 16);                // tdim1 hi=0; tile_dim0=32
  g1[4] = (int)rows;                                 // tile_dim1=rows; tile_dim2=0
  g1[5] = (int)strideElems;                          // tensor_dim0_stride[31:0]
  g1[6] = 0;                                         // stride0 hi, stride1 lo
  g1[7] = 0;
  v4i z4 = {0, 0, 0, 0};
#if defined(__clang_major__) && (__clang_major__ >= 23)
  v8i z8 = {0, 0, 0, 0, 0, 0, 0, 0};
  __builtin_amdgcn_tensor_load_to_lds(g0, g1, z4, z4, z8, 0);
#else
  __builtin_amdgcn_tensor_load_to_lds(g0, g1, z4, z4, 0);
#endif
}
#endif

// ---------------------------------------------------------------------------
// GEMM: C[M,N] = (Ah+Al)[M,K] * (Bh+Bl)[N,K]^T, bf16x3 WMMA, fp32 out.
// Block tile 128x64, 8 waves (4x2), wave tile 32x32 (2x2 frags).
// TDM path: double-buffered LDS, wave0 drives 4 descriptors per k-step.
// ---------------------------------------------------------------------------
__global__ __launch_bounds__(256)
void mla_gemm_bf16(const bf16_t* __restrict__ Ah, const bf16_t* __restrict__ Al,
                   const bf16_t* __restrict__ Bh, const bf16_t* __restrict__ Bl,
                   float* __restrict__ C, int M, int N, int K) {
#if MLA_USE_TDM
  __shared__ __align__(16) bf16_t sA[2][2][GTM * GPAD];
  __shared__ __align__(16) bf16_t sB[2][2][GTN * GPAD];
#else
  __shared__ __align__(16) bf16_t sA[1][2][GTM * GPAD];
  __shared__ __align__(16) bf16_t sB[1][2][GTN * GPAD];
#endif
  const int tid  = threadIdx.x;
  const int bm   = blockIdx.y * GTM;
  const int bn   = blockIdx.x * GTN;
  const int wave = tid >> 5, lane = tid & 31;
  const int lrow = lane & 15, half = lane >> 4;
  const int wm   = (wave >> 1) * 32;
  const int wn   = (wave & 1) * 32;
  const int nk   = K / GTK;

  v8f acc[2][2] = {};

#if MLA_USE_TDM
  if (wave == 0) {
    tdm_load_tile(sA[0][0], Ah + (size_t)bm * K, GTM, (unsigned)K);
    tdm_load_tile(sA[0][1], Al + (size_t)bm * K, GTM, (unsigned)K);
    tdm_load_tile(sB[0][0], Bh + (size_t)bn * K, GTN, (unsigned)K);
    tdm_load_tile(sB[0][1], Bl + (size_t)bn * K, GTN, (unsigned)K);
  }
#endif

  for (int t = 0; t < nk; ++t) {
#if MLA_USE_TDM
    const int cur = t & 1;
    if (wave == 0) {
      if (t + 1 < nk) {
        const int nb = cur ^ 1;
        const size_t ko = (size_t)(t + 1) * GTK;
        tdm_load_tile(sA[nb][0], Ah + (size_t)bm * K + ko, GTM, (unsigned)K);
        tdm_load_tile(sA[nb][1], Al + (size_t)bm * K + ko, GTM, (unsigned)K);
        tdm_load_tile(sB[nb][0], Bh + (size_t)bn * K + ko, GTN, (unsigned)K);
        tdm_load_tile(sB[nb][1], Bl + (size_t)bn * K + ko, GTN, (unsigned)K);
        __builtin_amdgcn_s_wait_tensorcnt(4);   // previous 4 tiles landed
      } else {
        __builtin_amdgcn_s_wait_tensorcnt(0);
      }
    }
    __syncthreads();
#else
    const int cur = 0;
    const int k0 = t * GTK;
    #pragma unroll
    for (int i = 0; i < 2; ++i) {               // A: 512 chunks of 8 bf16
      int idx = tid + i * 256;
      int r = idx >> 2, c8 = idx & 3;
      *(v8bf*)&sA[0][0][r * GPAD + c8 * 8] =
          *(const v8bf*)(Ah + (size_t)(bm + r) * K + k0 + c8 * 8);
      *(v8bf*)&sA[0][1][r * GPAD + c8 * 8] =
          *(const v8bf*)(Al + (size_t)(bm + r) * K + k0 + c8 * 8);
    }
    {                                            // B: 256 chunks
      int r = tid >> 2, c8 = tid & 3;
      *(v8bf*)&sB[0][0][r * GPAD + c8 * 8] =
          *(const v8bf*)(Bh + (size_t)(bn + r) * K + k0 + c8 * 8);
      *(v8bf*)&sB[0][1][r * GPAD + c8 * 8] =
          *(const v8bf*)(Bl + (size_t)(bn + r) * K + k0 + c8 * 8);
    }
    __syncthreads();
#endif

    const bf16_t* pAh = sA[cur][0];
    const bf16_t* pAl = sA[cur][1];
    const bf16_t* pBh = sB[cur][0];
    const bf16_t* pBl = sB[cur][1];
    const int ako = half * 8, bko = half * 8;
    v16bf ahf[2], alf[2], bhf[2], blf[2];
    #pragma unroll
    for (int i = 0; i < 2; ++i) {
      const bf16_t* p = &pAh[(wm + i * 16 + lrow) * GPAD + ako];
      ahf[i] = cat8(*(const v8bf*)p, *(const v8bf*)(p + 16));
      const bf16_t* q = &pAl[(wm + i * 16 + lrow) * GPAD + ako];
      alf[i] = cat8(*(const v8bf*)q, *(const v8bf*)(q + 16));
      const bf16_t* pb = &pBh[(wn + i * 16 + lrow) * GPAD + bko];
      bhf[i] = cat8(*(const v8bf*)pb, *(const v8bf*)(pb + 16));
      const bf16_t* qb = &pBl[(wn + i * 16 + lrow) * GPAD + bko];
      blf[i] = cat8(*(const v8bf*)qb, *(const v8bf*)(qb + 16));
    }
    #pragma unroll
    for (int i = 0; i < 2; ++i)
      #pragma unroll
      for (int j = 0; j < 2; ++j) {
        acc[i][j] = wmma_bf(ahf[i], bhf[j], acc[i][j]);
        acc[i][j] = wmma_bf(ahf[i], blf[j], acc[i][j]);
        acc[i][j] = wmma_bf(alf[i], bhf[j], acc[i][j]);
      }
    __syncthreads();
  }

  #pragma unroll
  for (int i = 0; i < 2; ++i)
    #pragma unroll
    for (int j = 0; j < 2; ++j)
      #pragma unroll
      for (int e = 0; e < 8; ++e) {
        int row = bm + wm + i * 16 + half * 8 + e;
        int col = bn + wn + j * 16 + lrow;
        C[(size_t)row * N + col] = acc[i][j][e];
      }
  (void)M;
}

// ---------------------------------------------------------------------------
// RMSNorm + RoPE on q (in place) and k (in place inside kv buffer).
// ---------------------------------------------------------------------------
__device__ __forceinline__ void norm_rope8(float* p, const float* __restrict__ w,
                                           int d0, const float* cosp,
                                           const float* sinp) {
  f32x4 a = *(const f32x4*)p;
  f32x4 b = *(const f32x4*)(p + 4);
  float ss = 0.f;
  #pragma unroll
  for (int e = 0; e < 4; ++e) ss += a[e] * a[e] + b[e] * b[e];
  ss = rsum16(ss);
  float inv = rsqrtf(ss * (1.0f / 128.0f) + 1e-6f);
  float v[8];
  #pragma unroll
  for (int e = 0; e < 4; ++e) {
    v[e]     = a[e] * inv * w[d0 + e];
    v[4 + e] = b[e] * inv * w[d0 + 4 + e];
  }
  float o[8];
  #pragma unroll
  for (int pr = 0; pr < 4; ++pr) {
    int i = (d0 >> 1) + pr;
    float c = cosp[i], s = sinp[i];
    float re = v[2 * pr], im = v[2 * pr + 1];
    o[2 * pr]     = re * c - im * s;
    o[2 * pr + 1] = re * s + im * c;
  }
  f32x4 oa = {o[0], o[1], o[2], o[3]};
  f32x4 ob = {o[4], o[5], o[6], o[7]};
  *(f32x4*)p       = oa;
  *(f32x4*)(p + 4) = ob;
}

__global__ __launch_bounds__(256)
void mla_qkprep(float* __restrict__ q, float* __restrict__ kvb,
                const float* __restrict__ qw, const float* __restrict__ kw,
                const float* __restrict__ fcos, const float* __restrict__ fsin) {
  const int t   = blockIdx.x;
  const int tid = threadIdx.x;
  const int h   = tid >> 4;
  const int d0  = (tid & 15) * 8;
  const float* cosp = fcos + (size_t)t * (HD / 2);
  const float* sinp = fsin + (size_t)t * (HD / 2);
  norm_rope8(q   + (size_t)t * DIMC       + h * HD + d0, qw, d0, cosp, sinp);
  norm_rope8(kvb + (size_t)t * (2 * DIMC) + h * HD + d0, kw, d0, cosp, sinp);
}

// ---------------------------------------------------------------------------
// Causal flash attention; K/V staged straight from pre-split bf16 hi/lo.
// ---------------------------------------------------------------------------
__global__ __launch_bounds__(256)
void mla_attn(const float* __restrict__ q, const bf16_t* __restrict__ kvh,
              const bf16_t* __restrict__ kvl, float* __restrict__ aout) {
  __shared__ __align__(16) bf16_t sKh[32 * 136], sKl[32 * 136];   // [kcol][d]
  __shared__ __align__(16) bf16_t sVh[128 * 40], sVl[128 * 40];   // [d][kcol]
  __shared__ __align__(16) bf16_t sPh[8 * 16 * 40], sPl[8 * 16 * 40];

  const int h    = blockIdx.x;
  const int qb   = blockIdx.y;
  const int tid  = threadIdx.x;
  const int wave = tid >> 5, lane = tid & 31;
  const int lrow = lane & 15, half = lane >> 4;
  const int qr0  = qb * 128 + wave * 16;
  const float SCALE = 0.08838834764831845f;   // 1/sqrt(128)
  const float NEG   = -3.0e38f;

  // Preload scaled Q frags (A layout, hi/lo) from fp32 q.
  v16bf qh[4], ql[4];
  {
    const float* qrow = q + (size_t)(qr0 + lrow) * DIMC + h * HD;
    const int ko = half * 8;
    #pragma unroll
    for (int c = 0; c < 4; ++c) {
      f32x4 x0 = *(const f32x4*)(qrow + c * 32 + ko);
      f32x4 x1 = *(const f32x4*)(qrow + c * 32 + ko + 4);
      f32x4 x2 = *(const f32x4*)(qrow + c * 32 + ko + 16);
      f32x4 x3 = *(const f32x4*)(qrow + c * 32 + ko + 20);
      v16bf hh, ll;
      #pragma unroll
      for (int e = 0; e < 4; ++e) {
        bf16_t hb, lb;
        splitbf(x0[e] * SCALE, hb, lb); hh[e]      = hb; ll[e]      = lb;
        splitbf(x1[e] * SCALE, hb, lb); hh[4 + e]  = hb; ll[4 + e]  = lb;
        splitbf(x2[e] * SCALE, hb, lb); hh[8 + e]  = hb; ll[8 + e]  = lb;
        splitbf(x3[e] * SCALE, hb, lb); hh[12 + e] = hb; ll[12 + e] = lb;
      }
      qh[c] = hh; ql[c] = ll;
    }
  }

  v8f o[8] = {};
  float mr[8], lr[8];
  #pragma unroll
  for (int e = 0; e < 8; ++e) { mr[e] = NEG; lr[e] = 0.f; }

  const int jend = qb * 128 + 128;
  for (int j0 = 0; j0 < jend; j0 += 32) {
    // Stage K rows (row-major) and V (transposed) from bf16 hi/lo.
    #pragma unroll
    for (int i = 0; i < 2; ++i) {
      int idx = tid + i * 256;               // 512 chunks of 8 bf16
      int r = idx >> 4, c8 = idx & 15;
      const size_t base = (size_t)(j0 + r) * (2 * DIMC) + h * HD + c8 * 8;
      *(v8bf*)&sKh[r * 136 + c8 * 8] = *(const v8bf*)(kvh + base);
      *(v8bf*)&sKl[r * 136 + c8 * 8] = *(const v8bf*)(kvl + base);
      v8bf vh = *(const v8bf*)(kvh + base + DIMC);
      v8bf vl = *(const v8bf*)(kvl + base + DIMC);
      #pragma unroll
      for (int e = 0; e < 8; ++e) {
        int d = c8 * 8 + e;
        sVh[d * 40 + r] = vh[e];
        sVl[d * 40 + r] = vl[e];
      }
    }
    __syncthreads();

    if (j0 <= qr0 + 15) {                    // wave-uniform: EXEC stays all-1
      v8f s0 = {}, s1 = {};
      #pragma unroll
      for (int c = 0; c < 4; ++c) {
        const bf16_t* p0h = &sKh[lrow * 136 + c * 32 + half * 8];
        const bf16_t* p0l = &sKl[lrow * 136 + c * 32 + half * 8];
        v16bf b0h = cat8(*(const v8bf*)p0h, *(const v8bf*)(p0h + 16));
        v16bf b0l = cat8(*(const v8bf*)p0l, *(const v8bf*)(p0l + 16));
        s0 = wmma_bf(qh[c], b0h, s0);
        s0 = wmma_bf(qh[c], b0l, s0);
        s0 = wmma_bf(ql[c], b0h, s0);
        const bf16_t* p1h = &sKh[(16 + lrow) * 136 + c * 32 + half * 8];
        const bf16_t* p1l = &sKl[(16 + lrow) * 136 + c * 32 + half * 8];
        v16bf b1h = cat8(*(const v8bf*)p1h, *(const v8bf*)(p1h + 16));
        v16bf b1l = cat8(*(const v8bf*)p1l, *(const v8bf*)(p1l + 16));
        s1 = wmma_bf(qh[c], b1h, s1);
        s1 = wmma_bf(qh[c], b1l, s1);
        s1 = wmma_bf(ql[c], b1h, s1);
      }

      float p0[8], p1[8], sc[8];
      #pragma unroll
      for (int e = 0; e < 8; ++e) {
        int grow = qr0 + half * 8 + e;
        float a0 = (j0 + lrow      > grow) ? NEG : s0[e];
        float a1 = (j0 + 16 + lrow > grow) ? NEG : s1[e];
        float rm = rmax16(fmaxf(a0, a1));
        float mn = fmaxf(mr[e], rm);
        sc[e] = __expf(mr[e] - mn);
        mr[e] = mn;
        p0[e] = __expf(a0 - mn);
        p1[e] = __expf(a1 - mn);
        lr[e] = lr[e] * sc[e] + rsum16(p0[e] + p1[e]);
      }
      #pragma unroll
      for (int g = 0; g < 8; ++g)
        #pragma unroll
        for (int e = 0; e < 8; ++e) o[g][e] *= sc[e];

      // C-layout P -> per-wave LDS -> A-layout frag.
      #pragma unroll
      for (int e = 0; e < 8; ++e) {
        int m = half * 8 + e;
        bf16_t hb, lb;
        splitbf(p0[e], hb, lb);
        sPh[wave * 640 + m * 40 + lrow] = hb;
        sPl[wave * 640 + m * 40 + lrow] = lb;
        splitbf(p1[e], hb, lb);
        sPh[wave * 640 + m * 40 + 16 + lrow] = hb;
        sPl[wave * 640 + m * 40 + 16 + lrow] = lb;
      }
      wave_lds_sync();
      const int ko = half * 8;
      const bf16_t* pa = &sPh[wave * 640 + lrow * 40 + ko];
      const bf16_t* pl = &sPl[wave * 640 + lrow * 40 + ko];
      v16bf pah = cat8(*(const v8bf*)pa, *(const v8bf*)(pa + 16));
      v16bf pal = cat8(*(const v8bf*)pl, *(const v8bf*)(pl + 16));

      #pragma unroll
      for (int g = 0; g < 8; ++g) {
        const bf16_t* vbh = &sVh[(g * 16 + lrow) * 40 + half * 8];
        const bf16_t* vbl = &sVl[(g * 16 + lrow) * 40 + half * 8];
        v16bf bvh = cat8(*(const v8bf*)vbh, *(const v8bf*)(vbh + 16));
        v16bf bvl = cat8(*(const v8bf*)vbl, *(const v8bf*)(vbl + 16));
        o[g] = wmma_bf(pah, bvh, o[g]);
        o[g] = wmma_bf(pah, bvl, o[g]);
        o[g] = wmma_bf(pal, bvh, o[g]);
      }
    }
    __syncthreads();
  }

  float invl[8];
  #pragma unroll
  for (int e = 0; e < 8; ++e) invl[e] = 1.0f / lr[e];
  #pragma unroll
  for (int g = 0; g < 8; ++g)
    #pragma unroll
    for (int e = 0; e < 8; ++e) {
      int row = qr0 + half * 8 + e;
      aout[(size_t)row * DIMC + h * HD + g * 16 + lrow] = o[g][e] * invl[e];
    }
}

// ---------------------------------------------------------------------------
extern "C" void kernel_launch(void* const* d_in, const int* in_sizes, int n_in,
                              void* d_out, int out_size, void* d_ws, size_t ws_size,
                              hipStream_t stream) {
  const float* x    = (const float*)d_in[0];
  const float* wq   = (const float*)d_in[1];
  const float* wkvd = (const float*)d_in[2];
  const float* wkvu = (const float*)d_in[3];
  const float* wo   = (const float*)d_in[4];
  const float* qnw  = (const float*)d_in[5];
  const float* knw  = (const float*)d_in[6];
  const float* fcos = (const float*)d_in[7];
  const float* fsin = (const float*)d_in[8];
  float* out = (float*)d_out;

  const size_t E_x   = (size_t)SEQ * DIMC;
  const size_t E_w   = (size_t)DIMC * DIMC;
  const size_t E_kd  = (size_t)KVR * DIMC;
  const size_t E_ku  = (size_t)2 * DIMC * KVR;
  const size_t E_lat = (size_t)SEQ * KVR;
  const size_t E_kv  = (size_t)SEQ * 2 * DIMC;

  char* w = (char*)d_ws;
  auto alloc = [&](size_t bytes) {
    void* p = (void*)w;
    w += (bytes + 255) & ~(size_t)255;
    return p;
  };
  bf16_t* xh    = (bf16_t*)alloc(E_x * 2);
  bf16_t* xl    = (bf16_t*)alloc(E_x * 2);
  bf16_t* wqh   = (bf16_t*)alloc(E_w * 2);
  bf16_t* wql   = (bf16_t*)alloc(E_w * 2);
  bf16_t* wkvdh = (bf16_t*)alloc(E_kd * 2);
  bf16_t* wkvdl = (bf16_t*)alloc(E_kd * 2);
  bf16_t* wkvuh = (bf16_t*)alloc(E_ku * 2);
  bf16_t* wkvul = (bf16_t*)alloc(E_ku * 2);
  bf16_t* woh   = (bf16_t*)alloc(E_w * 2);
  bf16_t* wol   = (bf16_t*)alloc(E_w * 2);
  float*  xq    = (float*)alloc(E_x * 4);
  float*  lat   = (float*)alloc(E_lat * 4);
  bf16_t* lath  = (bf16_t*)alloc(E_lat * 2);
  bf16_t* latl  = (bf16_t*)alloc(E_lat * 2);
  float*  kvb   = (float*)alloc(E_kv * 4);
  bf16_t* kvh   = (bf16_t*)alloc(E_kv * 2);
  bf16_t* kvl   = (bf16_t*)alloc(E_kv * 2);
  float*  attn  = (float*)alloc(E_x * 4);
  bf16_t* ath   = (bf16_t*)alloc(E_x * 2);
  bf16_t* atl   = (bf16_t*)alloc(E_x * 2);

  dim3 blk(256, 1, 1);
  auto splitN = [&](const float* s, bf16_t* h, bf16_t* l, size_t n) {
    int n4 = (int)(n / 4);
    mla_split<<<dim3((n4 + 255) / 256), blk, 0, stream>>>(s, h, l, n4);
  };

  // One-time operand splits (all L2-resident; ~5us of traffic total).
  splitN(x, xh, xl, E_x);
  splitN(wq, wqh, wql, E_w);
  splitN(wkvd, wkvdh, wkvdl, E_kd);
  splitN(wkvu, wkvuh, wkvul, E_ku);
  splitN(wo, woh, wol, E_w);

  // xq = x * wq^T
  mla_gemm_bf16<<<dim3(DIMC / GTN, SEQ / GTM), blk, 0, stream>>>(
      xh, xl, wqh, wql, xq, SEQ, DIMC, DIMC);
  // latent = x * w_kv_down^T
  mla_gemm_bf16<<<dim3(KVR / GTN, SEQ / GTM), blk, 0, stream>>>(
      xh, xl, wkvdh, wkvdl, lat, SEQ, KVR, DIMC);
  splitN(lat, lath, latl, E_lat);
  // kv = latent * w_kv_up^T
  mla_gemm_bf16<<<dim3((2 * DIMC) / GTN, SEQ / GTM), blk, 0, stream>>>(
      lath, latl, wkvuh, wkvul, kvb, SEQ, 2 * DIMC, KVR);
  // RMSNorm + RoPE on q and k (in place), then split kv for attention.
  mla_qkprep<<<dim3(SEQ), blk, 0, stream>>>(xq, kvb, qnw, knw, fcos, fsin);
  splitN(kvb, kvh, kvl, E_kv);
  // causal flash attention
  mla_attn<<<dim3(NH, SEQ / 128), blk, 0, stream>>>(xq, kvh, kvl, attn);
  splitN(attn, ath, atl, E_x);
  // out = attn * wo^T
  mla_gemm_bf16<<<dim3(DIMC / GTN, SEQ / GTM), blk, 0, stream>>>(
      ath, atl, woh, wol, out, SEQ, DIMC, DIMC);

  (void)in_sizes; (void)n_in; (void)out_size; (void)ws_size;
}
